// ConceptGenerator_53515292508969
// MI455X (gfx1250) — compile-verified, matrix operand order checked by reading, not run on hardware
//
#include <hip/hip_runtime.h>
#include <hip/hip_bf16.h>
#include <math.h>

typedef unsigned short u16;
typedef __attribute__((ext_vector_type(16))) __bf16 v16bf;
typedef __attribute__((ext_vector_type(8)))  float  v8f;
typedef __attribute__((ext_vector_type(8)))  unsigned short us8;
typedef __attribute__((ext_vector_type(16))) unsigned short us16;

#define T_  128
#define B_  32
#define S_  512
#define E_  512
#define FF_ 2048
#define C_  512
#define V_  12000

__device__ __forceinline__ u16 f2bf(float f) {
  unsigned u = __builtin_bit_cast(unsigned, f);
  unsigned r = u + 0x7FFFu + ((u >> 16) & 1u);   // RNE (NaN edge ignored)
  return (u16)(r >> 16);
}

// Load a 16-element bf16 fragment chunk per CDNA5 16-bit A/B layout:
// p points at (row, kb + 8*hiHalf); elements 0..7 -> +0..7, 8..15 -> +16..23
__device__ __forceinline__ v16bf load_frag(const u16* p) {
  us8 lo = *(const us8*)(p);
  us8 hi = *(const us8*)(p + 16);
  us16 u;
#pragma unroll
  for (int e = 0; e < 8; ++e) { u[e] = lo[e]; u[e + 8] = hi[e]; }
  return __builtin_bit_cast(v16bf, u);
}

// ---------------------------------------------------------------- converts
__global__ void __launch_bounds__(256) f32_to_bf16_k(const float* __restrict__ x,
                                                     u16* __restrict__ y, long n) {
  long i = (long)blockIdx.x * 256 + threadIdx.x;
  if (i < n) y[i] = f2bf(x[i]);
}

__global__ void __launch_bounds__(64) zero_f32_k(float* p, int n) {
  int i = blockIdx.x * 64 + threadIdx.x;
  if (i < n) p[i] = 0.0f;
}

// v[(s*B+b)*E+e]  ->  vt[((b*E)+e)*S+s]   (make K=s contiguous for attn GEMM)
__global__ void __launch_bounds__(256) transpose_v_k(const u16* __restrict__ v,
                                                     u16* __restrict__ vt) {
  long i = (long)blockIdx.x * 256 + threadIdx.x;
  if (i >= (long)S_ * B_ * E_) return;
  long s = i / ((long)B_ * E_);
  long rem = i - s * (long)B_ * E_;
  long b = rem / E_, e = rem - b * E_;
  vt[((long)b * E_ + e) * S_ + s] = v[i];
}

// ---------------------------------------------------------------- WMMA GEMM
// C[m,n] = act( (sum_k A[m,k]*B[n,k] + bias[n]) * scale )
// A: bf16 rows k-contiguous (row stride lda);  B: bf16 rows k-contiguous (row stride ldb)
// One wave per 16(M) x 64(N) tile: 1 A fragment reused by 4 WMMAs per K-step.
// act: 0 none, 1 relu, 2 tanh
__global__ void __launch_bounds__(32)
gemm_bf16(const u16* __restrict__ A, long lda, long sAb,
          const u16* __restrict__ Bm, long ldb, long sBb,
          const float* __restrict__ bias,
          float* __restrict__ Cf, long sCfb,
          u16* __restrict__ Cb, long sCbb,
          long ldc, int K, float scale, int act) {
  const int lane = threadIdx.x;
  const int hiq = lane >> 4, nl = lane & 15;
  const long m0 = (long)blockIdx.y * 16, n0 = (long)blockIdx.x * 64;
  const long bz = blockIdx.z;
  const u16* Ab = A + bz * sAb + (m0 + nl) * lda + 8 * hiq;
  const u16* Bb = Bm + bz * sBb + (n0 + nl) * ldb + 8 * hiq;
  v8f cacc[4] = {};
  for (int kb = 0; kb < K; kb += 32) {
    if (kb + 32 < K) {                      // speculative prefetch of next chunk
      __builtin_prefetch(Ab + kb + 32, 0, 1);
      __builtin_prefetch(Bb + kb + 32, 0, 1);
    }
    v16bf a = load_frag(Ab + kb);
#pragma unroll
    for (int j = 0; j < 4; ++j) {
      v16bf b = load_frag(Bb + (long)j * 16 * ldb + kb);
      cacc[j] = __builtin_amdgcn_wmma_f32_16x16x32_bf16(false, a, false, b,
                                                        (short)0, cacc[j],
                                                        false, false);
    }
  }
#pragma unroll
  for (int j = 0; j < 4; ++j) {
    const long n = n0 + j * 16 + nl;
    const float bv = bias ? bias[n] : 0.0f;
#pragma unroll
    for (int i = 0; i < 8; ++i) {
      const long m = m0 + i + 8 * hiq;
      float v = (cacc[j][i] + bv) * scale;
      if (act == 1) v = fmaxf(v, 0.0f);
      else if (act == 2) v = tanhf(v);
      const long off = m * ldc + n;
      if (Cf) Cf[bz * sCfb + off] = v;
      if (Cb) Cb[bz * sCbb + off] = f2bf(v);
    }
  }
}

// ---------------------------------------------------------------- softmax over S
__global__ void __launch_bounds__(128)
softmax_mask_k(const float* __restrict__ scores, const unsigned char* __restrict__ mask,
               float* __restrict__ wOut, u16* __restrict__ wBf) {
  const int t = blockIdx.x, b = blockIdx.y, tid = threadIdx.x;
  const float* row = scores + ((long)b * T_ + t) * S_;
  const unsigned char* mrow = mask + (long)b * S_;
  __shared__ float red[4];
  float vals[4];
  float mx = -INFINITY;
#pragma unroll
  for (int j = 0; j < 4; ++j) {
    int s = tid + j * 128;
    float v = mrow[s] ? -INFINITY : row[s];
    vals[j] = v;
    mx = fmaxf(mx, v);
  }
  for (int off = 16; off > 0; off >>= 1) mx = fmaxf(mx, __shfl_xor(mx, off, 32));
  if ((tid & 31) == 0) red[tid >> 5] = mx;
  __syncthreads();
  mx = fmaxf(fmaxf(red[0], red[1]), fmaxf(red[2], red[3]));
  float sum = 0.0f;
#pragma unroll
  for (int j = 0; j < 4; ++j) {
    vals[j] = (vals[j] == -INFINITY) ? 0.0f : expf(vals[j] - mx);
    sum += vals[j];
  }
  for (int off = 16; off > 0; off >>= 1) sum += __shfl_xor(sum, off, 32);
  __syncthreads();
  if ((tid & 31) == 0) red[tid >> 5] = sum;
  __syncthreads();
  sum = red[0] + red[1] + red[2] + red[3];
  const float inv = sum > 0.0f ? 1.0f / sum : 0.0f;
#pragma unroll
  for (int j = 0; j < 4; ++j) {
    int s = tid + j * 128;
    float wv = vals[j] * inv;
    wOut[((long)t * B_ + b) * S_ + s] = wv;          // alignment_weight (T,B,S)
    wBf[((long)b * T_ + t) * S_ + s] = f2bf(wv);     // per-batch (T,S) bf16 for GEMM
  }
}

// ---------------------------------------------------------------- add + layernorm
__device__ __forceinline__ float block_sum128(float v, float* red, int tid) {
  for (int off = 16; off > 0; off >>= 1) v += __shfl_xor(v, off, 32);
  __syncthreads();
  if ((tid & 31) == 0) red[tid >> 5] = v;
  __syncthreads();
  return red[0] + red[1] + red[2] + red[3];
}

__global__ void __launch_bounds__(128)
add_ln_k(const float* __restrict__ X, const float* __restrict__ R,
         const float* __restrict__ g, const float* __restrict__ be,
         float* __restrict__ outF, u16* __restrict__ outB) {
  const long r = blockIdx.x;
  const int tid = threadIdx.x;
  __shared__ float red[4];
  const float* x = X + r * E_;
  const float* rr = R + r * E_;
  float v[4];
  float s1 = 0.0f;
#pragma unroll
  for (int j = 0; j < 4; ++j) {
    int e = tid + j * 128;
    v[j] = x[e] + rr[e];
    s1 += v[j];
  }
  const float mean = block_sum128(s1, red, tid) * (1.0f / E_);
  float s2 = 0.0f;
#pragma unroll
  for (int j = 0; j < 4; ++j) { float d = v[j] - mean; s2 += d * d; }
  __syncthreads();
  const float rstd = rsqrtf(block_sum128(s2, red, tid) * (1.0f / E_) + 1e-5f);
#pragma unroll
  for (int j = 0; j < 4; ++j) {
    int e = tid + j * 128;
    float o = (v[j] - mean) * rstd * g[e] + be[e];
    if (outF) outF[r * E_ + e] = o;
    if (outB) outB[r * E_ + e] = f2bf(o);
  }
}

// ---------------------------------------------------------------- 3-way gates
__global__ void __launch_bounds__(32)
gates_k(const float* __restrict__ oc, const float* __restrict__ Wd,
        const float* __restrict__ bd, float* __restrict__ gates) {
  const long r = blockIdx.x;
  const int lane = threadIdx.x;
  const float* row = oc + r * C_;
  float d0 = 0, d1 = 0, d2 = 0;
  for (int e = lane; e < C_; e += 32) {
    float x = row[e];
    d0 += x * Wd[e];
    d1 += x * Wd[C_ + e];
    d2 += x * Wd[2 * C_ + e];
  }
  for (int off = 16; off > 0; off >>= 1) {
    d0 += __shfl_xor(d0, off, 32);
    d1 += __shfl_xor(d1, off, 32);
    d2 += __shfl_xor(d2, off, 32);
  }
  if (lane == 0) {
    d0 += bd[0]; d1 += bd[1]; d2 += bd[2];
    float m = fmaxf(d0, fmaxf(d1, d2));
    float e0 = expf(d0 - m), e1 = expf(d1 - m), e2 = expf(d2 - m);
    float inv = 1.0f / (e0 + e1 + e2);
    gates[r * 3 + 0] = e0 * inv;
    gates[r * 3 + 1] = e1 * inv;
    gates[r * 3 + 2] = e2 * inv;
  }
}

// ---------------------------------------------------------------- fused vocab logsumexp + loss
// 16 rows of oc cached in LDS; 4 waves stream Wg tiles through WMMA with online logsumexp.
__global__ void __launch_bounds__(128)
loss_k(const u16* __restrict__ ocb, const u16* __restrict__ Wgb,
       const float* __restrict__ bg, const float* __restrict__ gates,
       const int* __restrict__ target, const int* __restrict__ copy_seq,
       const float* __restrict__ wAlign, float* __restrict__ lossOut) {
  __shared__ __align__(16) u16 soc[16 * C_];
  __shared__ float stgtlog[16];
  __shared__ int stgt[16];
  __shared__ float pmax[4][16];
  __shared__ float psum[4][16];
  __shared__ float cpart[128];

  const int tid = threadIdx.x;
  const long r0 = (long)blockIdx.x * 16;

  for (int idx = tid; idx < 16 * C_; idx += 128) soc[idx] = ocb[r0 * C_ + idx];
  if (tid < 16) { stgt[tid] = target[r0 + tid]; stgtlog[tid] = 0.0f; }
  __syncthreads();

  const int wave = tid >> 5, lane = tid & 31;
  const int hiq = lane >> 4, nl = lane & 15;
  float mx[8], sm[8];
#pragma unroll
  for (int i = 0; i < 8; ++i) { mx[i] = -INFINITY; sm[i] = 0.0f; }

  const u16* ap = soc + nl * C_ + 8 * hiq;
  for (int tile = wave; tile < V_ / 16; tile += 4) {
    const int n0 = tile * 16;
    const u16* bp = Wgb + (long)(n0 + nl) * C_ + 8 * hiq;
    v8f c = {};
#pragma unroll
    for (int kb = 0; kb < C_; kb += 32) {
      v16bf a = load_frag(ap + kb);                 // LDS fragment (ds_load_b128)
      v16bf b = load_frag(bp + kb);                 // global fragment
      c = __builtin_amdgcn_wmma_f32_16x16x32_bf16(false, a, false, b, (short)0, c,
                                                  false, false);
    }
    const int vv = n0 + nl;
    const float bgv = bg[vv];
#pragma unroll
    for (int i = 0; i < 8; ++i) {
      const int m = i + 8 * hiq;
      float lg = c[i] + bgv;
      if (vv == stgt[m]) stgtlog[m] = lg;           // exactly one lane matches
      float M = fmaxf(mx[i], lg);
      sm[i] = sm[i] * expf(mx[i] - M) + expf(lg - M);
      mx[i] = M;
    }
  }
  // merge the 16 lanes holding each row (within each half of the wave)
#pragma unroll
  for (int off = 1; off < 16; off <<= 1) {
#pragma unroll
    for (int i = 0; i < 8; ++i) {
      float omx = __shfl_xor(mx[i], off, 32);
      float osm = __shfl_xor(sm[i], off, 32);
      float M = fmaxf(mx[i], omx);
      sm[i] = sm[i] * expf(mx[i] - M) + osm * expf(omx - M);
      mx[i] = M;
    }
  }
  if (nl == 0) {
#pragma unroll
    for (int i = 0; i < 8; ++i) {
      pmax[wave][i + 8 * hiq] = mx[i];
      psum[wave][i + 8 * hiq] = sm[i];
    }
  }
  __syncthreads();

  // copy-mechanism contribution: 8 threads per row scan S
  {
    const int i = tid & 15, sub = tid >> 4;
    const long r = r0 + i;
    const int t = (int)(r / B_), b = (int)(r % B_);
    const int tg = stgt[i];
    const float g1 = gates[r * 3 + 1], g2 = gates[r * 3 + 2];
    float acc = 0.0f;
    for (int s = sub; s < S_; s += 8) {
      float wv = wAlign[((long)t * B_ + b) * S_ + s];
      int c0 = copy_seq[((long)s * B_ + b) * 2 + 0];
      int c1 = copy_seq[((long)s * B_ + b) * 2 + 1];
      acc += wv * ((c0 == tg ? g2 : 0.0f) + (c1 == tg ? g1 : 0.0f));
    }
    cpart[tid] = acc;
  }
  __syncthreads();

  if (tid < 16) {
    float cp = 0.0f;
    for (int sb = 0; sb < 8; ++sb) cp += cpart[sb * 16 + tid];
    float M = pmax[0][tid];
    for (int w = 1; w < 4; ++w) M = fmaxf(M, pmax[w][tid]);
    float SM = 0.0f;
    for (int w = 0; w < 4; ++w) SM += psum[w][tid] * expf(pmax[w][tid] - M);
    const long r = r0 + tid;
    const float g0 = gates[r * 3 + 0];
    const float pgen = g0 * expf(stgtlog[tid] - M) / SM;
    const float p = pgen + cp;
    const int b = (int)(r % B_);
    const float loss = (stgt[tid] == 0) ? 0.0f : -logf(p + 1e-12f);
    atomicAdd(&lossOut[b], loss);
  }
}

// ---------------------------------------------------------------- host launch
extern "C" void kernel_launch(void* const* d_in, const int* in_sizes, int n_in,
                              void* d_out, int out_size, void* d_ws, size_t ws_size,
                              hipStream_t stream) {
  (void)in_sizes; (void)n_in; (void)out_size; (void)ws_size;
  const float* outs = (const float*)d_in[0];
  const float* snt  = (const float*)d_in[1];
  const unsigned char* mask = (const unsigned char*)d_in[2];
  const int* copy_seq = (const int*)d_in[3];
  const int* target   = (const int*)d_in[4];
  const float* Wq = (const float*)d_in[6];  const float* bq = (const float*)d_in[7];
  const float* Wk = (const float*)d_in[8];  const float* bk = (const float*)d_in[9];
  const float* Wv = (const float*)d_in[10]; const float* bv = (const float*)d_in[11];
  const float* Wo = (const float*)d_in[12]; const float* bo = (const float*)d_in[13];
  const float* ln1_g = (const float*)d_in[14]; const float* ln1_b = (const float*)d_in[15];
  const float* W1 = (const float*)d_in[16]; const float* b1 = (const float*)d_in[17];
  const float* W2 = (const float*)d_in[18]; const float* b2 = (const float*)d_in[19];
  const float* ln2_g = (const float*)d_in[20]; const float* ln2_b = (const float*)d_in[21];
  const float* Wt = (const float*)d_in[22]; const float* bt = (const float*)d_in[23];
  const float* Wg = (const float*)d_in[24]; const float* bg = (const float*)d_in[25];
  const float* Wd = (const float*)d_in[26]; const float* bd = (const float*)d_in[27];

  const long TB = (long)T_ * B_;    // 4096
  const long SB = (long)S_ * B_;    // 16384

  char* ws = (char*)d_ws;
  size_t off = 0;
  auto alloc = [&](size_t bytes) -> char* {
    char* p = ws + off;
    off += (bytes + 255) & ~(size_t)255;
    return p;
  };
  u16* outsbf = (u16*)alloc(TB * E_ * 2);
  u16* sntbf  = (u16*)alloc(SB * E_ * 2);
  u16* Wqbf = (u16*)alloc((long)E_ * E_ * 2);
  u16* Wkbf = (u16*)alloc((long)E_ * E_ * 2);
  u16* Wvbf = (u16*)alloc((long)E_ * E_ * 2);
  u16* Wobf = (u16*)alloc((long)E_ * E_ * 2);
  u16* W1bf = (u16*)alloc((long)FF_ * E_ * 2);
  u16* W2bf = (u16*)alloc((long)E_ * FF_ * 2);
  u16* Wtbf = (u16*)alloc((long)C_ * E_ * 2);
  u16* Wgbf = (u16*)alloc((long)V_ * C_ * 2);
  u16* qbf  = (u16*)alloc(TB * E_ * 2);
  u16* kbf  = (u16*)alloc(SB * E_ * 2);
  u16* vbf  = (u16*)alloc(SB * E_ * 2);
  u16* vtbf = (u16*)alloc((long)B_ * E_ * S_ * 2);
  float* scores = (float*)alloc((long)B_ * T_ * S_ * 4);
  u16* wbf    = (u16*)alloc((long)B_ * T_ * S_ * 2);
  u16* attnbf = (u16*)alloc(TB * E_ * 2);
  float* x1f  = (float*)alloc(TB * E_ * 4);
  float* ln1f = (float*)alloc(TB * E_ * 4);
  u16* ln1bf  = (u16*)alloc(TB * E_ * 2);
  u16* hbf    = (u16*)alloc(TB * FF_ * 2);
  float* x2f  = (float*)alloc(TB * E_ * 4);
  u16* o2bf   = (u16*)alloc(TB * E_ * 2);
  float* ocf  = (float*)alloc(TB * C_ * 4);
  u16* ocbf   = (u16*)alloc(TB * C_ * 2);
  float* gates = (float*)alloc(TB * 3 * 4);

  float* lossOut  = (float*)d_out;
  float* outs2Out = (float*)d_out + B_;
  float* alignOut = (float*)d_out + B_ + TB * E_;

  zero_f32_k<<<1, 64, 0, stream>>>(lossOut, B_);

  auto cvt = [&](const float* src, u16* dst, long n) {
    f32_to_bf16_k<<<(unsigned)((n + 255) / 256), 256, 0, stream>>>(src, dst, n);
  };
  cvt(outs, outsbf, TB * E_);
  cvt(snt, sntbf, SB * E_);
  cvt(Wq, Wqbf, (long)E_ * E_);
  cvt(Wk, Wkbf, (long)E_ * E_);
  cvt(Wv, Wvbf, (long)E_ * E_);
  cvt(Wo, Wobf, (long)E_ * E_);
  cvt(W1, W1bf, (long)FF_ * E_);
  cvt(W2, W2bf, (long)E_ * FF_);
  cvt(Wt, Wtbf, (long)C_ * E_);
  cvt(Wg, Wgbf, (long)V_ * C_);

  const float scale = 1.0f / sqrtf((float)E_);

  // q = (outs @ Wq^T + bq) * scale   -> bf16
  gemm_bf16<<<dim3(E_ / 64, TB / 16, 1), 32, 0, stream>>>(
      outsbf, E_, 0, Wqbf, E_, 0, bq, (float*)nullptr, 0, qbf, 0, E_, E_, scale, 0);
  // k, v
  gemm_bf16<<<dim3(E_ / 64, SB / 16, 1), 32, 0, stream>>>(
      sntbf, E_, 0, Wkbf, E_, 0, bk, (float*)nullptr, 0, kbf, 0, E_, E_, 1.0f, 0);
  gemm_bf16<<<dim3(E_ / 64, SB / 16, 1), 32, 0, stream>>>(
      sntbf, E_, 0, Wvbf, E_, 0, bv, (float*)nullptr, 0, vbf, 0, E_, E_, 1.0f, 0);
  transpose_v_k<<<(unsigned)((SB * E_ + 255) / 256), 256, 0, stream>>>(vbf, vtbf);

  // scores[b] (T,S) = q_b @ k_b^T  (batched over b)
  gemm_bf16<<<dim3(S_ / 64, T_ / 16, B_), 32, 0, stream>>>(
      qbf, (long)B_ * E_, E_, kbf, (long)B_ * E_, E_, (const float*)nullptr,
      scores, (long)T_ * S_, (u16*)nullptr, 0, S_, E_, 1.0f, 0);

  softmax_mask_k<<<dim3(T_, B_), 128, 0, stream>>>(scores, mask, alignOut, wbf);

  // attn[t,b,:] = w_b @ v_b  (batched, v transposed so K=s contiguous)
  gemm_bf16<<<dim3(E_ / 64, T_ / 16, B_), 32, 0, stream>>>(
      wbf, S_, (long)T_ * S_, vtbf, S_, (long)E_ * S_, (const float*)nullptr,
      (float*)nullptr, 0, attnbf, E_, (long)B_ * E_, S_, 1.0f, 0);

  // x1 = attn @ Wo^T + bo
  gemm_bf16<<<dim3(E_ / 64, TB / 16, 1), 32, 0, stream>>>(
      attnbf, E_, 0, Wobf, E_, 0, bo, x1f, 0, (u16*)nullptr, 0, E_, E_, 1.0f, 0);
  add_ln_k<<<(unsigned)TB, 128, 0, stream>>>(outs, x1f, ln1_g, ln1_b, ln1f, ln1bf);

  // FFN
  gemm_bf16<<<dim3(FF_ / 64, TB / 16, 1), 32, 0, stream>>>(
      ln1bf, E_, 0, W1bf, E_, 0, b1, (float*)nullptr, 0, hbf, 0, FF_, E_, 1.0f, 1);
  gemm_bf16<<<dim3(E_ / 64, TB / 16, 1), 32, 0, stream>>>(
      hbf, FF_, 0, W2bf, FF_, 0, b2, x2f, 0, (u16*)nullptr, 0, E_, FF_, 1.0f, 0);
  add_ln_k<<<(unsigned)TB, 128, 0, stream>>>(ln1f, x2f, ln2_g, ln2_b, outs2Out, o2bf);

  // oc = tanh(outs2 @ Wt^T + bt)
  gemm_bf16<<<dim3(C_ / 64, TB / 16, 1), 32, 0, stream>>>(
      o2bf, E_, 0, Wtbf, E_, 0, bt, ocf, 0, ocbf, 0, C_, E_, 1.0f, 2);

  gates_k<<<(unsigned)TB, 32, 0, stream>>>(ocf, Wd, bd, gates);

  loss_k<<<(unsigned)(TB / 16), 128, 0, stream>>>(ocbf, Wgbf, bg, gates, target,
                                                  copy_seq, alignOut, lossOut);
}
